// ATOC_90280212562557
// MI455X (gfx1250) — matrix operand order repaired and checked by reading, not compiled, not run on hardware
//
#include <hip/hip_runtime.h>

// ---------------------------------------------------------------------------
// Types for CDNA5 WMMA
// ---------------------------------------------------------------------------
typedef __attribute__((ext_vector_type(16))) __bf16         v16bf;
typedef __attribute__((ext_vector_type(16))) unsigned short v16us;
typedef __attribute__((ext_vector_type(8)))  unsigned short v8us;
typedef __attribute__((ext_vector_type(8)))  float          v8f;

#define D_IN  128
#define D_H   256
#define D_C   32
#define D_OUT 64

// f32 -> bf16 round-to-nearest-even
__device__ __forceinline__ unsigned short f2bf(float f) {
    unsigned int u = __builtin_bit_cast(unsigned int, f);
    u += 0x7fffu + ((u >> 16) & 1u);
    return (unsigned short)(u >> 16);
}
__device__ __forceinline__ float bf2f(unsigned short u) {
    return __builtin_bit_cast(float, ((unsigned int)u) << 16);
}

// ---------------------------------------------------------------------------
// Elementwise f32 -> bf16 (for activations, row-major kept as-is)
// ---------------------------------------------------------------------------
__global__ void k_f32_to_bf16(const float* __restrict__ in,
                              unsigned short* __restrict__ out, int n) {
    int i = blockIdx.x * blockDim.x + threadIdx.x;
    if (i < n) out[i] = f2bf(in[i]);
}

// Weight convert + pack into fragment-major layout:
//   Bp[((k/32)*Nc + n)*32 + (k%32)] = bf16(W[k*Nc + n])
// so each lane's 16-element B fragment (K = k0 + e + 16*hi, fixed n) is one
// contiguous, 32B-aligned v16us load.
__global__ void k_pack_w_bf16(const float* __restrict__ W,
                              unsigned short* __restrict__ Bp, int K, int Nc) {
    int i = blockIdx.x * blockDim.x + threadIdx.x;
    if (i >= K * Nc) return;
    int k = i / Nc, n = i % Nc;
    Bp[((size_t)(k >> 5) * Nc + n) * 32 + (k & 31)] = f2bf(W[i]);
}

__global__ void k_zero_u32(unsigned int* __restrict__ p, int n) {
    int i = blockIdx.x * blockDim.x + threadIdx.x;
    if (i < n) p[i] = 0u;
}

// ---------------------------------------------------------------------------
// Tiled bf16 WMMA GEMM:  C[M,Nc] = A[M,K] @ W[K,Nc] + bias  (opt. ReLU)
// A row-major bf16 (lda halves); Bp fragment-major packed bf16 (see above);
// C f32 or bf16 (flags bit1), ReLU (flags bit0).
// One wave32 computes a 64x32 strip (4 M-tiles x 2 N-tiles, 8 WMMAs/K-step).
// GVS addressing: uniform scalar pointer advances over K (s_add), lane part
// is a loop-invariant 32-bit offset -> no per-iteration VALU address math and
// no WMMA->VALU hazard NOPs in the hot loop.
// Requires M % 16 == 0, K % 32 == 0, Nc % 32 == 0.
// ---------------------------------------------------------------------------
#define GEMM_TM 4
#define GEMM_TN 2

__global__ __launch_bounds__(256)
void k_gemm_wmma_bf16(const unsigned short* __restrict__ A, int lda,
                      const unsigned short* __restrict__ Bp,
                      const float* __restrict__ bias,
                      void* __restrict__ Cv, int ldc,
                      int M, int K, int Nc, int flags) {
    const int lane = threadIdx.x & 31;
    const int wid  = (blockIdx.x * blockDim.x + threadIdx.x) >> 5;
    const int tilesN = Nc / (16 * GEMM_TN);
    const int tilesM = (M + 16 * GEMM_TM - 1) / (16 * GEMM_TM);
    if (wid >= tilesM * tilesN) return;          // wave-uniform exit

    const int tm = wid / tilesN;
    const int tn = wid % tilesN;
    const int m0 = tm * 16 * GEMM_TM;
    const int n0 = tn * 16 * GEMM_TN;
    const int l15 = lane & 15;
    const int hi  = lane >> 4;                   // 0: lanes 0-15, 1: lanes 16-31

    // Lane-divergent, loop-invariant 32-bit offsets (in halves).
    int mbase[GEMM_TM];
    int aoff[GEMM_TM];
#pragma unroll
    for (int mi = 0; mi < GEMM_TM; ++mi) {
        int mb = m0 + 16 * mi;
        if (mb > M - 16) mb = M - 16;            // wave-uniform clamp
        mbase[mi] = mb;
        aoff[mi]  = (mb + l15) * lda + hi * 16;
    }
    int boff[GEMM_TN];
#pragma unroll
    for (int ni = 0; ni < GEMM_TN; ++ni)
        boff[ni] = (n0 + 16 * ni + l15) * 32 + hi * 16;

    v8f acc[GEMM_TM][GEMM_TN];
#pragma unroll
    for (int mi = 0; mi < GEMM_TM; ++mi)
#pragma unroll
        for (int ni = 0; ni < GEMM_TN; ++ni)
#pragma unroll
            for (int r = 0; r < 8; ++r) acc[mi][ni][r] = 0.0f;

    // Uniform scalar pointers, advanced by scalar adds each K-step.
    const unsigned short* Ak = A;                // advances by 32 halves
    const unsigned short* Bk = Bp;               // advances by Nc*32 halves
    const int bpanel = Nc * 32;

    for (int k0 = 0; k0 < K; k0 += 32) {
        v16bf bfrag[GEMM_TN];
#pragma unroll
        for (int ni = 0; ni < GEMM_TN; ++ni) {
            const v16us bu = *(const v16us*)(Bk + boff[ni]);
            bfrag[ni] = __builtin_bit_cast(v16bf, bu);
        }
#pragma unroll
        for (int mi = 0; mi < GEMM_TM; ++mi) {
            // A fragment: two contiguous 8-half runs per lane.
            const v8us a_lo = *(const v8us*)(Ak + aoff[mi]);
            const v8us a_hi = *(const v8us*)(Ak + aoff[mi] + 8);
            const v16us au = __builtin_shufflevector(
                a_lo, a_hi, 0, 1, 2, 3, 4, 5, 6, 7, 8, 9, 10, 11, 12, 13, 14, 15);
            const v16bf afrag = __builtin_bit_cast(v16bf, au);
#pragma unroll
            for (int ni = 0; ni < GEMM_TN; ++ni) {
                acc[mi][ni] = __builtin_amdgcn_wmma_f32_16x16x32_bf16(
                    false, afrag, false, bfrag[ni],
                    (short)0, acc[mi][ni], false, false);
            }
        }
        Ak += 32;
        Bk += bpanel;
    }

    const bool relu  = (flags & 1) != 0;
    const bool bf16o = (flags & 2) != 0;
    float* Cf          = (float*)Cv;
    unsigned short* Cb = (unsigned short*)Cv;

    float bsv[GEMM_TN];
#pragma unroll
    for (int ni = 0; ni < GEMM_TN; ++ni) bsv[ni] = bias[n0 + 16 * ni + l15];

#pragma unroll
    for (int mi = 0; mi < GEMM_TM; ++mi) {
        if (m0 + 16 * mi >= M) continue;         // only clamped duplicates skipped
#pragma unroll
        for (int ni = 0; ni < GEMM_TN; ++ni) {
            const int col = n0 + 16 * ni + l15;
#pragma unroll
            for (int r = 0; r < 8; ++r) {
                float vv = acc[mi][ni][r] + bsv[ni];
                if (relu) vv = fmaxf(vv, 0.0f);
                const int row = mbase[mi] + r + hi * 8;    // C layout: M = r + 8*hi
                if (bf16o) Cb[(size_t)row * ldc + col] = f2bf(vv);
                else       Cf[(size_t)row * ldc + col] = vv;
            }
        }
    }
}

// ---------------------------------------------------------------------------
// Edge phase: dedupe (faithful to mask .set overwrite), CSR build, softmax+gather
// ---------------------------------------------------------------------------
__global__ void k_edge_claim(const int* __restrict__ ei, unsigned int* __restrict__ bm,
                             int* __restrict__ cnt, int* __restrict__ flag,
                             int E, int N) {
    int e = blockIdx.x * blockDim.x + threadIdx.x;
    if (e >= E) return;
    int i = ei[e], j = ei[E + e];
    unsigned long long bitpos = (unsigned long long)i * (unsigned)N + (unsigned)j;
    unsigned int word = (unsigned int)(bitpos >> 5);
    unsigned int bit  = 1u << (bitpos & 31u);
    unsigned int old  = atomicOr(&bm[word], bit);
    if (!(old & bit)) { flag[e] = 1; atomicAdd(&cnt[i], 1); }
    else              { flag[e] = 0; }
}

// Single-block exclusive scan of cnt[0..n) -> rowoff[0..n], rowoff[n]=total.
__global__ void k_scan(const int* __restrict__ cnt, int* __restrict__ rowoff,
                       int* __restrict__ cursor, int n) {
    __shared__ int buf[1024];
    __shared__ int carry;
    const int tid = threadIdx.x;
    if (tid == 0) carry = 0;
    __syncthreads();
    for (int base = 0; base < n; base += 1024) {
        int idx = base + tid;
        int v = (idx < n) ? cnt[idx] : 0;
        buf[tid] = v;
        __syncthreads();
        for (int off = 1; off < 1024; off <<= 1) {
            int t = (tid >= off) ? buf[tid - off] : 0;
            __syncthreads();
            buf[tid] += t;
            __syncthreads();
        }
        int incl = buf[tid];
        int total = buf[1023];
        int excl = carry + incl - v;
        if (idx < n) { rowoff[idx] = excl; cursor[idx] = excl; }
        __syncthreads();
        if (tid == 0) carry += total;
        __syncthreads();
    }
    if (tid == 0) rowoff[n] = carry;
}

__global__ void k_edge_fill(const int* __restrict__ ei, const int* __restrict__ flag,
                            int* __restrict__ cursor, int* __restrict__ colbuf, int E) {
    int e = blockIdx.x * blockDim.x + threadIdx.x;
    if (e >= E || !flag[e]) return;
    int i = ei[e], j = ei[E + e];
    int pos = atomicAdd(&cursor[i], 1);
    colbuf[pos] = j;
}

// One wave32 per row: softmax over neighbors (the +1.0 edge bias cancels
// exactly inside softmax) and comm[i] = sum_j w_ij * v[j]. Writes bf16 into
// the right half of the combined buffer. Rows with no edges -> zeros
// (matches nan_to_num on the all -inf softmax row).
__global__ void k_row_attn(const float* __restrict__ q, const float* __restrict__ kx,
                           const unsigned short* __restrict__ v,
                           const int* __restrict__ rowoff, const int* __restrict__ colbuf,
                           unsigned short* __restrict__ comb, int n) {
    const int lane = threadIdx.x & 31;
    const int row  = (blockIdx.x * blockDim.x + threadIdx.x) >> 5;
    if (row >= n) return;                         // wave-uniform
    const float scale = 0.17677669529663687f;     // 1/sqrt(32)

    const float qv = q[row * D_C + lane];
    const int s0 = rowoff[row], s1 = rowoff[row + 1];

    float rowsum = 0.0f;
    for (int p = s0; p < s1; ++p) {
        int j = colbuf[p];
        float s = qv * kx[j * D_C + lane];
#pragma unroll
        for (int off = 16; off; off >>= 1) s += __shfl_xor(s, off, 32);
        rowsum += __expf(s * scale);              // same value in all lanes
    }
    const float inv = (rowsum > 0.0f) ? (1.0f / rowsum) : 0.0f;

    float acc[D_H / 32];
#pragma unroll
    for (int t = 0; t < D_H / 32; ++t) acc[t] = 0.0f;

    for (int p = s0; p < s1; ++p) {
        int j = colbuf[p];
        float s = qv * kx[j * D_C + lane];
#pragma unroll
        for (int off = 16; off; off >>= 1) s += __shfl_xor(s, off, 32);
        float w = __expf(s * scale) * inv;
#pragma unroll
        for (int t = 0; t < D_H / 32; ++t)
            acc[t] += w * bf2f(v[(size_t)j * D_H + lane + 32 * t]);
    }
#pragma unroll
    for (int t = 0; t < D_H / 32; ++t)
        comb[(size_t)row * (2 * D_H) + D_H + lane + 32 * t] = f2bf(acc[t]);
}

// ---------------------------------------------------------------------------
// Host orchestration
// ---------------------------------------------------------------------------
static inline int ceil_div(int a, int b) { return (a + b - 1) / b; }

static void launch_gemm(const unsigned short* A, int lda,
                        const unsigned short* Bp,
                        const float* bias, void* C, int ldc,
                        int M, int K, int Nc, int flags, hipStream_t s) {
    int tilesN = Nc / (16 * GEMM_TN);
    int tilesM = ceil_div(M, 16 * GEMM_TM);
    int waves  = tilesM * tilesN;
    int blocks = ceil_div(waves, 8);              // 8 waves (256 threads) / block
    k_gemm_wmma_bf16<<<blocks, 256, 0, s>>>(A, lda, Bp, bias, C, ldc, M, K, Nc, flags);
}

extern "C" void kernel_launch(void* const* d_in, const int* in_sizes, int n_in,
                              void* d_out, int out_size, void* d_ws, size_t ws_size,
                              hipStream_t stream) {
    const float* x    = (const float*)d_in[0];
    const int*   ei   = (const int*)  d_in[1];
    const float* W_in = (const float*)d_in[2];
    const float* b_in = (const float*)d_in[3];
    const float* Wq   = (const float*)d_in[4];
    const float* bq   = (const float*)d_in[5];
    const float* Wk   = (const float*)d_in[6];
    const float* bk   = (const float*)d_in[7];
    const float* Wv   = (const float*)d_in[8];
    const float* bv   = (const float*)d_in[9];
    const float* W1   = (const float*)d_in[10];
    const float* b1   = (const float*)d_in[11];
    const float* W2   = (const float*)d_in[12];
    const float* b2   = (const float*)d_in[13];

    const int N = in_sizes[0] / D_IN;             // 10000
    const int E = in_sizes[1] / 2;                // 320000

    // ---- workspace carve-up (256B aligned) ----
    char* ws = (char*)d_ws;
    size_t off = 0;
    auto carve = [&](size_t bytes) -> char* {
        char* p = ws + off;
        off = (off + bytes + 255) & ~(size_t)255;
        return p;
    };
    unsigned short* comb  = (unsigned short*)carve((size_t)N * 2 * D_H * 2); // bf16 [N,512]
    float*          qbuf  = (float*)carve((size_t)N * D_C * 4);
    float*          kbuf  = (float*)carve((size_t)N * D_C * 4);
    unsigned short* vbuf  = (unsigned short*)carve((size_t)N * D_H * 2);
    unsigned short* m1buf = (unsigned short*)carve((size_t)N * D_H * 2);
    unsigned short* xb    = (unsigned short*)carve((size_t)N * D_IN * 2);
    unsigned short* wb    = (unsigned short*)carve((size_t)(2 * D_H) * D_H * 2); // max weight
    const size_t bmWords  = ((size_t)N * (size_t)N + 31) / 32;
    unsigned int*   bm    = (unsigned int*)carve(bmWords * 4);
    int*            cnt   = (int*)carve((size_t)N * 4);
    int*            cur   = (int*)carve((size_t)N * 4);
    int*            roff  = (int*)carve((size_t)(N + 1) * 4);
    int*            flag  = (int*)carve((size_t)E * 4);
    int*            col   = (int*)carve((size_t)E * 4);
    (void)ws_size;

    // ---- 1. h = x @ W_in + b_in  (bf16 out -> comb[:, 0:256]) ----
    k_f32_to_bf16<<<ceil_div(N * D_IN, 256), 256, 0, stream>>>(x, xb, N * D_IN);
    k_pack_w_bf16<<<ceil_div(D_IN * D_H, 256), 256, 0, stream>>>(W_in, wb, D_IN, D_H);
    launch_gemm(xb, D_IN, wb, b_in, comb, 2 * D_H, N, D_IN, D_H, /*bf16out*/2, stream);

    // ---- 2. q, k (f32 out), v (bf16 out), all reading h from comb ----
    k_pack_w_bf16<<<ceil_div(D_H * D_C, 256), 256, 0, stream>>>(Wq, wb, D_H, D_C);
    launch_gemm(comb, 2 * D_H, wb, bq, qbuf, D_C, N, D_H, D_C, 0, stream);
    k_pack_w_bf16<<<ceil_div(D_H * D_C, 256), 256, 0, stream>>>(Wk, wb, D_H, D_C);
    launch_gemm(comb, 2 * D_H, wb, bk, kbuf, D_C, N, D_H, D_C, 0, stream);
    k_pack_w_bf16<<<ceil_div(D_H * D_H, 256), 256, 0, stream>>>(Wv, wb, D_H, D_H);
    launch_gemm(comb, 2 * D_H, wb, bv, vbuf, D_H, N, D_H, D_H, 2, stream);

    // ---- 3. sparse attention: dedupe -> CSR -> per-row softmax gather ----
    k_zero_u32<<<ceil_div((int)bmWords, 256), 256, 0, stream>>>(bm, (int)bmWords);
    k_zero_u32<<<ceil_div(N, 256), 256, 0, stream>>>((unsigned int*)cnt, N);
    k_edge_claim<<<ceil_div(E, 256), 256, 0, stream>>>(ei, bm, cnt, flag, E, N);
    k_scan<<<1, 1024, 0, stream>>>(cnt, roff, cur, N);
    k_edge_fill<<<ceil_div(E, 256), 256, 0, stream>>>(ei, flag, cur, col, E);
    k_row_attn<<<ceil_div(N * 32, 256), 256, 0, stream>>>(qbuf, kbuf, vbuf, roff, col,
                                                          comb, N);

    // ---- 4. MLP: relu(comb @ W1 + b1) @ W2 + b2 ----
    k_pack_w_bf16<<<ceil_div(2 * D_H * D_H, 256), 256, 0, stream>>>(W1, wb, 2 * D_H, D_H);
    launch_gemm(comb, 2 * D_H, wb, b1, m1buf, D_H, N, 2 * D_H, D_H, /*relu|bf16*/3, stream);
    k_pack_w_bf16<<<ceil_div(D_H * D_OUT, 256), 256, 0, stream>>>(W2, wb, D_H, D_OUT);
    launch_gemm(m1buf, D_H, wb, b2, d_out, D_OUT, N, D_H, D_OUT, 0, stream);
    (void)out_size; (void)n_in;
}
